// LlamaSdpaAttention_24842090840154
// MI455X (gfx1250) — matrix operand order repaired and checked by reading, not compiled, not run on hardware
//
#include <hip/hip_runtime.h>
#include <hip/hip_bf16.h>

// ---------------- problem constants (from reference) ----------------
#define B_    2
#define S_    2048
#define D_    4096
#define NH_   32
#define NKV_  8
#define HD_   128
#define KVD_  (NKV_ * HD_)        // 1024
#define NTOT_ (D_ + 2 * KVD_)     // 6144 fused QKV output columns

typedef __bf16 bf16;
typedef __attribute__((ext_vector_type(16))) __bf16 v16bf;
typedef __attribute__((ext_vector_type(8)))  float  v8f;

#define DEV static __device__ __forceinline__

// D = A(16x32 bf16) * B(32x16 bf16) + C(16x16 f32), wave32
DEV v8f wmma_bf16(v16bf a, v16bf b, v8f c) {
    return __builtin_amdgcn_wmma_f32_16x16x32_bf16(
        /*neg_a=*/false, a, /*neg_b=*/false, b,
        /*c_mod=*/(short)0, c, /*reuse_a=*/false, /*reuse_b=*/false);
}

// A-fragment (16x32, 16-bit): lane l<16 holds row M=l, K=0..7 and 16..23;
// lanes 16..31 hold row M=l-16, K=8..15 and 24..31.
DEV v16bf load_a_f32(const float* __restrict__ src, int ld, int lane) {
    const int m = lane & 15, half = lane >> 4;
    const float* p0 = src + (size_t)m * ld + half * 8;
    const float* p1 = p0 + 16;
    v16bf a;
#pragma unroll
    for (int i = 0; i < 8; ++i) { a[i] = (bf16)p0[i]; a[8 + i] = (bf16)p1[i]; }
    return a;
}

DEV v16bf load_a_bf16(const bf16* __restrict__ src, int ld, int lane) {
    const int m = lane & 15, half = lane >> 4;
    const bf16* p0 = src + (size_t)m * ld + half * 8;
    const bf16* p1 = p0 + 16;
    v16bf a;
#pragma unroll
    for (int i = 0; i < 8; ++i) { a[i] = p0[i]; a[8 + i] = p1[i]; }
    return a;
}

// B-fragment (32x16, 16-bit): lane l<16 holds column N=l, K=0..7 and 16..23;
// lanes 16..31 hold column N=l-16, K=8..15 and 24..31.  "base" points at B[k0][n0] with
// column n stored contiguously in K (i.e. row n of the K-major source matrix).
// Works for both global and LDS-resident sources (addrspace is inferred).
DEV v16bf load_b_bf16(const bf16* base, int ld, int lane) {
    const int n = lane & 15, half = lane >> 4;
    const bf16* p = base + (size_t)n * ld + (half << 3);
    v16bf b;
#pragma unroll
    for (int i = 0; i < 16; ++i) b[i] = p[i + (i & 8)];
    return b;
}

// B-fragment where each lane already resolved its own weight row pointer
// (row_k0 = &W[n][k0]); lanes 16..31 read K offset +8 (K 8..15, 24..31).
DEV v16bf load_b_f32row(const float* __restrict__ row_k0, int lane) {
    const float* p = row_k0 + ((lane >> 4) << 3);
    v16bf b;
#pragma unroll
    for (int i = 0; i < 16; ++i) b[i] = (bf16)p[i + (i & 8)];
    return b;
}

// LDS byte offset of a generic pointer into __shared__: flat->LDS mapping
// uses addr[31:0] (ISA 10.2 aperture rules), so the low 32 bits are the
// wave-relative LDS byte address the async-DMA VDST operand wants.
DEV unsigned lds_byte_off(const void* p) { return (unsigned)(size_t)p; }

// One 32x128 bf16 K tile (contiguous 8 KB in memory) -> LDS, via 16
// GLOBAL_LOAD_ASYNC_TO_LDS_B128 ops (512 B per op across the wave).
DEV void stage_k_tile_async(const bf16* gk, unsigned lds_base, int lane) {
    const bf16* g = gk + lane * 8;                       // 16 B per lane
    const unsigned l = lds_base + (unsigned)lane * 16u;
#pragma unroll
    for (int c = 0; c < 16; ++c) {
        asm volatile("global_load_async_to_lds_b128 %0, %1, off"
                     :: "v"(l + (unsigned)c * 512u), "v"(g + c * 256)
                     : "memory");
    }
}

// =====================================================================
// Kernel 1: fused QKV projection.  C[M=B*S, N=6144] = x * [Wq;Wk;Wv]^T
// Wave: 32x64 tile (2x4 accumulators: 8 WMMA per 6 fragment loads).
// Block: 8 waves -> 128x128 tile.  Epilogue scatters into Q[b,h,s,d],
// K[b,h,s,d], Vt[b,h,d,s] as bf16.
// =====================================================================
__global__ __launch_bounds__(256) void qkv_gemm_kernel(
    const float* __restrict__ x,
    const float* __restrict__ Wq, const float* __restrict__ Wk,
    const float* __restrict__ Wv,
    bf16* __restrict__ Qb, bf16* __restrict__ Kb, bf16* __restrict__ Vt) {

    const int lane = threadIdx.x & 31;
    const int w    = threadIdx.x >> 5;
    const int rowW = blockIdx.x * 128 + (w >> 1) * 32;
    const int colW = blockIdx.y * 128 + (w & 1) * 64;
    const int n = lane & 15, hi = lane >> 4;

    // per-lane weight row pointers for the four N-subtiles
    const float* wrow[4];
#pragma unroll
    for (int j = 0; j < 4; ++j) {
        int gc = colW + j * 16 + n;
        if (gc < D_)              wrow[j] = Wq + (size_t)gc * D_;
        else if (gc < D_ + KVD_)  wrow[j] = Wk + (size_t)(gc - D_) * D_;
        else                      wrow[j] = Wv + (size_t)(gc - D_ - KVD_) * D_;
    }

    v8f acc[2][4];
#pragma unroll
    for (int i = 0; i < 2; ++i)
#pragma unroll
        for (int j = 0; j < 4; ++j)
#pragma unroll
            for (int r = 0; r < 8; ++r) acc[i][j][r] = 0.0f;

    for (int k0 = 0; k0 < D_; k0 += 32) {
        v16bf a0 = load_a_f32(x + (size_t)rowW * D_ + k0, D_, lane);
        v16bf a1 = load_a_f32(x + (size_t)(rowW + 16) * D_ + k0, D_, lane);
        v16bf bfr[4];
#pragma unroll
        for (int j = 0; j < 4; ++j) {
            bfr[j] = load_b_f32row(wrow[j] + k0, lane);
            if (k0 + 512 < D_) __builtin_prefetch(wrow[j] + k0 + 512, 0, 1);
        }
#pragma unroll
        for (int j = 0; j < 4; ++j) {
            acc[0][j] = wmma_bf16(a0, bfr[j], acc[0][j]);
            acc[1][j] = wmma_bf16(a1, bfr[j], acc[1][j]);
        }
    }

#pragma unroll
    for (int i = 0; i < 2; ++i)
#pragma unroll
        for (int j = 0; j < 4; ++j) {
            const int gc = colW + j * 16 + n;
#pragma unroll
            for (int r = 0; r < 8; ++r) {
                const int gm = rowW + i * 16 + r + hi * 8;   // token index
                const int bb = gm >> 11;                     // S_ == 2048
                const int s  = gm & (S_ - 1);
                const float f = acc[i][j][r];
                if (gc < D_) {
                    const int h = gc >> 7, d = gc & (HD_ - 1);
                    Qb[(((size_t)(bb * NH_ + h)) * S_ + s) * HD_ + d] = (bf16)f;
                } else if (gc < D_ + KVD_) {
                    const int c = gc - D_;
                    const int h = c >> 7, d = c & (HD_ - 1);
                    Kb[(((size_t)(bb * NKV_ + h)) * S_ + s) * HD_ + d] = (bf16)f;
                } else {
                    const int c = gc - D_ - KVD_;
                    const int h = c >> 7, d = c & (HD_ - 1);
                    Vt[(((size_t)(bb * NKV_ + h)) * HD_ + d) * S_ + s] = (bf16)f;
                }
            }
        }
}

// =====================================================================
// Kernel 2: RoPE applied in-place to Q and K (pairs d, d+64).
// =====================================================================
__global__ __launch_bounds__(256) void rope_kernel(bf16* __restrict__ Qb,
                                                   bf16* __restrict__ Kb) {
    const size_t total = (size_t)B_ * (NH_ + NKV_) * S_ * (HD_ / 2);
    size_t t = (size_t)blockIdx.x * blockDim.x + threadIdx.x;
    if (t >= total) return;
    const int d = (int)(t & 63);
    size_t rest = t >> 6;
    const int s  = (int)(rest & (S_ - 1));
    const int bh = (int)(rest >> 11);

    bf16* p;
    if (bh < B_ * NH_) p = Qb + ((size_t)bh * S_ + s) * HD_;
    else               p = Kb + ((size_t)(bh - B_ * NH_) * S_ + s) * HD_;

    // inv_freq = base^(-d/64),  base = 500000
    const float ang = (float)s * __powf(500000.0f, -(float)d * (1.0f / 64.0f));
    float si, c;
    __sincosf(ang, &si, &c);
    const float x0 = (float)p[d];
    const float x1 = (float)p[d + 64];
    p[d]      = (bf16)(x0 * c - x1 * si);
    p[d + 64] = (bf16)(x1 * c + x0 * si);
}

// =====================================================================
// Kernel 3: causal GQA flash attention.  One wave owns 16 query rows.
// K tiles are DMA'd into a double-buffered LDS stage with
// global_load_async_to_lds_b128 (ASYNCcnt), overlapping the next tile's
// DMA with this tile's compute.  Per 32-key step: 8 WMMA for Q*K^T
// (B-operand from LDS), online softmax (16-lane shfl reductions), P is
// bounced through a wave-private LDS slab (DScnt) to convert C-layout ->
// A-layout, then 8 WMMA for P*V (V pre-transposed, direct global).
// =====================================================================
__global__ __launch_bounds__(256) void attn_kernel(
    const bf16* __restrict__ Qb, const bf16* __restrict__ Kb,
    const bf16* __restrict__ Vt, bf16* __restrict__ ctx) {

    __shared__ bf16 kst[8][2][32 * 128];   // per-wave double-buffered K stage (128 KB)
    __shared__ bf16 pst[8][16][32];        // per-wave P tile (8 KB)

    const int lane = threadIdx.x & 31;
    const int w    = threadIdx.x >> 5;
    const int n = lane & 15, hi = lane >> 4;

    const int tile = blockIdx.x * 8 + w;          // [0, B*NH*S/16)
    const int q0 = (tile & (S_ / 16 - 1)) * 16;   // S_/16 == 128
    const int bh = tile >> 7;
    const int b = bh >> 5;                        // NH_ == 32
    const int h = bh & 31;
    const int kvh = h >> 2;                       // GQA 4x

    const bf16* Qh = Qb + (size_t)(b * NH_ + h) * S_ * HD_;
    const bf16* Kh = Kb + (size_t)(b * NKV_ + kvh) * S_ * HD_;
    const bf16* Vh = Vt + (size_t)(b * NKV_ + kvh) * HD_ * S_;

    const unsigned koff[2] = { lds_byte_off(&kst[w][0][0]),
                               lds_byte_off(&kst[w][1][0]) };

    // Q tile as 4 A-fragments over the HD=128 reduction dim
    v16bf qf[4];
#pragma unroll
    for (int kd = 0; kd < 4; ++kd)
        qf[kd] = load_a_bf16(Qh + (size_t)q0 * HD_ + kd * 32, HD_, lane);

    v8f o[8];
    float mrow[8], lrow[8];
#pragma unroll
    for (int r = 0; r < 8; ++r) {
        mrow[r] = -__builtin_inff();
        lrow[r] = 0.0f;
#pragma unroll
        for (int f = 0; f < 8; ++f) o[f][r] = 0.0f;
    }

    // 1/sqrt(128) * log2(e): softmax in exp2 domain
    const float sc = 0.08838834764831845f * 1.4426950408889634f;

    const int kend = q0 + 16;

    // prologue: DMA the first K tile into buffer 0
    stage_k_tile_async(Kh, koff[0], lane);

    for (int k0 = 0; k0 < kend; k0 += 32) {
        const int buf = (k0 >> 5) & 1;
        const bf16* kbuf = &kst[w][buf][0];

        // all older LDS reads must retire before overwriting the other buffer
        asm volatile("s_wait_dscnt 0" ::: "memory");
        const int nb = k0 + 32;
        if (nb < kend) {
            stage_k_tile_async(Kh + (size_t)nb * HD_, koff[buf ^ 1], lane);
            // current buffer ready; leave the 16 just-issued DMAs in flight
            asm volatile("s_wait_asynccnt 0x10" ::: "memory");
        } else {
            asm volatile("s_wait_asynccnt 0x0" ::: "memory");
        }

        // scores: Q (regs) x K^T (LDS stage)
        v8f s0, s1;
#pragma unroll
        for (int r = 0; r < 8; ++r) { s0[r] = 0.0f; s1[r] = 0.0f; }
#pragma unroll
        for (int kd = 0; kd < 4; ++kd) {
            v16bf bk0 = load_b_bf16(kbuf + kd * 32, HD_, lane);
            v16bf bk1 = load_b_bf16(kbuf + 16 * HD_ + kd * 32, HD_, lane);
            s0 = wmma_bf16(qf[kd], bk0, s0);
            s1 = wmma_bf16(qf[kd], bk1, s1);
        }

        float alpha[8];
#pragma unroll
        for (int r = 0; r < 8; ++r) {
            const int row = q0 + r + hi * 8;
            float v0 = (k0 + n      <= row) ? s0[r] * sc : -3.0e38f;
            float v1 = (k0 + 16 + n <= row) ? s1[r] * sc : -3.0e38f;
            // row max across the 16 lanes holding this row
            float t = fmaxf(v0, v1);
#pragma unroll
            for (int off = 1; off < 16; off <<= 1)
                t = fmaxf(t, __shfl_xor(t, off, 32));
            const float mnew = fmaxf(mrow[r], t);
            const float a  = exp2f(mrow[r] - mnew);
            const float p0 = exp2f(v0 - mnew);
            const float p1 = exp2f(v1 - mnew);
            float ps = p0 + p1;
#pragma unroll
            for (int off = 1; off < 16; off <<= 1)
                ps += __shfl_xor(ps, off, 32);
            lrow[r] = lrow[r] * a + ps;
            mrow[r] = mnew;
            alpha[r] = a;
            // spill P (C-layout) into the wave's LDS slab, row-major 16x32
            const int m = r + hi * 8;
            pst[w][m][n]      = (bf16)p0;
            pst[w][m][16 + n] = (bf16)p1;
        }

        // rescale running output
#pragma unroll
        for (int f = 0; f < 8; ++f)
#pragma unroll
            for (int r = 0; r < 8; ++r) o[f][r] *= alpha[r];

        // DS is in-order per wave, but make the store->load dep explicit
        asm volatile("s_wait_dscnt 0" ::: "memory");

        // reload P as an A-fragment (16x32 bf16)
        v16bf pf;
        {
            const bf16* pr = &pst[w][n][0];   // A row M == lane & 15 == n
            const int kof = hi * 8;
#pragma unroll
            for (int i = 0; i < 8; ++i) {
                pf[i]     = pr[kof + i];
                pf[8 + i] = pr[16 + kof + i];
            }
        }

        // O += P * V   (B operand from transposed V: contiguous in k)
#pragma unroll
        for (int f = 0; f < 8; ++f) {
            v16bf bv = load_b_bf16(Vh + (size_t)(f * 16) * S_ + k0, S_, lane);
            o[f] = wmma_bf16(pf, bv, o[f]);
        }
    }

    // normalize and store context as bf16 in [B*S, D] token-major layout
#pragma unroll
    for (int r = 0; r < 8; ++r) {
        const float inv = 1.0f / lrow[r];
        const int s = q0 + r + hi * 8;
        bf16* dst = ctx + ((size_t)(b * S_ + s)) * D_ + h * HD_;
#pragma unroll
        for (int f = 0; f < 8; ++f)
            dst[f * 16 + n] = (bf16)(o[f][r] * inv);
    }
}

// =====================================================================
// Kernel 4: output projection.  out[B*S, D] = ctx(bf16) * Wo^T(f32)
// Wave: 32x64 tile (2x4 accumulators).  Block: 8 waves -> 128x128 tile.
// =====================================================================
__global__ __launch_bounds__(256) void oproj_gemm_kernel(
    const bf16* __restrict__ ctx, const float* __restrict__ Wo,
    float* __restrict__ out) {

    const int lane = threadIdx.x & 31;
    const int w    = threadIdx.x >> 5;
    const int rowW = blockIdx.x * 128 + (w >> 1) * 32;
    const int colW = blockIdx.y * 128 + (w & 1) * 64;
    const int n = lane & 15, hi = lane >> 4;

    const float* wrow[4];
#pragma unroll
    for (int j = 0; j < 4; ++j)
        wrow[j] = Wo + (size_t)(colW + j * 16 + n) * D_;

    v8f acc[2][4];
#pragma unroll
    for (int i = 0; i < 2; ++i)
#pragma unroll
        for (int j = 0; j < 4; ++j)
#pragma unroll
            for (int r = 0; r < 8; ++r) acc[i][j][r] = 0.0f;

    for (int k0 = 0; k0 < D_; k0 += 32) {
        v16bf a0 = load_a_bf16(ctx + (size_t)rowW * D_ + k0, D_, lane);
        v16bf a1 = load_a_bf16(ctx + (size_t)(rowW + 16) * D_ + k0, D_, lane);
        v16bf bfr[4];
#pragma unroll
        for (int j = 0; j < 4; ++j) {
            bfr[j] = load_b_f32row(wrow[j] + k0, lane);
            if (k0 + 512 < D_) __builtin_prefetch(wrow[j] + k0 + 512, 0, 1);
        }
#pragma unroll
        for (int j = 0; j < 4; ++j) {
            acc[0][j] = wmma_bf16(a0, bfr[j], acc[0][j]);
            acc[1][j] = wmma_bf16(a1, bfr[j], acc[1][j]);
        }
    }

#pragma unroll
    for (int i = 0; i < 2; ++i)
#pragma unroll
        for (int j = 0; j < 4; ++j)
#pragma unroll
            for (int r = 0; r < 8; ++r) {
                const int gm = rowW + i * 16 + r + hi * 8;
                const int gc = colW + j * 16 + n;
                out[(size_t)gm * D_ + gc] = acc[i][j][r];
            }
}

// =====================================================================
extern "C" void kernel_launch(void* const* d_in, const int* in_sizes, int n_in,
                              void* d_out, int out_size, void* d_ws, size_t ws_size,
                              hipStream_t stream) {
    (void)in_sizes; (void)n_in; (void)out_size; (void)ws_size;

    const float* x  = (const float*)d_in[0];
    const float* Wq = (const float*)d_in[1];
    const float* Wk = (const float*)d_in[2];
    const float* Wv = (const float*)d_in[3];
    const float* Wo = (const float*)d_in[4];
    float* out = (float*)d_out;

    // workspace carve-out (bf16 intermediates, ~80 MB total)
    char* p = (char*)d_ws;
    bf16* Qb  = (bf16*)p;  p += (size_t)B_ * NH_  * S_ * HD_ * sizeof(bf16);  // 32 MB
    bf16* Kb  = (bf16*)p;  p += (size_t)B_ * NKV_ * S_ * HD_ * sizeof(bf16);  //  8 MB
    bf16* Vt  = (bf16*)p;  p += (size_t)B_ * NKV_ * HD_ * S_ * sizeof(bf16);  //  8 MB
    bf16* ctx = (bf16*)p;                                                     // 32 MB

    // 1) fused QKV projection (WMMA bf16, f32 accumulate)
    dim3 g1((B_ * S_) / 128, NTOT_ / 128);
    qkv_gemm_kernel<<<g1, 256, 0, stream>>>(x, Wq, Wk, Wv, Qb, Kb, Vt);

    // 2) RoPE on Q and K
    const size_t pairs = (size_t)B_ * (NH_ + NKV_) * S_ * (HD_ / 2);
    rope_kernel<<<(unsigned)((pairs + 255) / 256), 256, 0, stream>>>(Qb, Kb);

    // 3) causal GQA flash attention (async-DMA K staging)
    attn_kernel<<<(B_ * NH_ * (S_ / 16)) / 8, 256, 0, stream>>>(Qb, Kb, Vt, ctx);

    // 4) output projection
    dim3 g4((B_ * S_) / 128, D_ / 128);
    oproj_gemm_kernel<<<g4, 256, 0, stream>>>(ctx, Wo, out);
}
